// ChunkIndexer_67654324847338
// MI455X (gfx1250) — matrix operand order repaired and checked by reading, not compiled, not run on hardware
//
#include <hip/hip_runtime.h>
#include <hip/hip_bf16.h>

#define BATCH       2048
#define NUM_CHUNKS  65536
#define EMBED       512
#define TOPK        16
#define QTILE       32                       // queries per WG
#define NSPLIT      64
#define SPLIT_CHUNKS (NUM_CHUNKS / NSPLIT)   // 1024 chunks per WG
#define NPW         32                       // chunks per wave (2 N-frags)
#define NBLOCK      256                      // chunks per WG iteration (8 waves * 32)
#define NB_ITERS    (SPLIT_CHUNKS / NBLOCK)  // 4
#define TPQ         8                        // scan threads per query (256/QTILE)
#define SB_STRIDE   130                      // 128 cols + pad (bank-safe scan)

typedef __bf16 bf16_t;
typedef __attribute__((ext_vector_type(16))) __bf16 v16bf;
typedef __attribute__((ext_vector_type(8)))  __bf16 v8bf;
typedef __attribute__((ext_vector_type(8)))  float   v8f;

// ---------------- f32 -> bf16 (round to nearest even) ----------------
__device__ __forceinline__ unsigned short f32_to_bf16_rne(float f) {
    unsigned u = __float_as_uint(f);
    unsigned r = u + 0x7FFFu + ((u >> 16) & 1u);
    return (unsigned short)(r >> 16);
}

__global__ void __launch_bounds__(256)
cvt_bf16_kernel(const float* __restrict__ src, unsigned short* __restrict__ dst, int n4) {
    int i = blockIdx.x * blockDim.x + threadIdx.x;
    if (i >= n4) return;
    float4 f = ((const float4*)src)[i];
    ushort4 o;
    o.x = f32_to_bf16_rne(f.x);
    o.y = f32_to_bf16_rne(f.y);
    o.z = f32_to_bf16_rne(f.z);
    o.w = f32_to_bf16_rne(f.w);
    ((ushort4*)dst)[i] = o;
}

// ------- sorted top-k insertion (static indices only -> stays in VGPRs) -------
#define TOPK_INIT(S, I_)                                      \
    _Pragma("unroll")                                         \
    for (int _j = 0; _j < TOPK; ++_j) { S[_j] = -3.0e38f; I_[_j] = 0; }

#define TOPK_INSERT(S, I_, v, id)                             \
    do {                                                      \
        float _v = (v); int _id = (id);                       \
        if (_v > S[TOPK - 1]) {                               \
            S[TOPK - 1] = _v; I_[TOPK - 1] = _id;             \
            _Pragma("unroll")                                 \
            for (int _j = TOPK - 1; _j > 0; --_j) {           \
                if (S[_j] > S[_j - 1]) {                      \
                    float _ts = S[_j]; S[_j] = S[_j-1]; S[_j-1] = _ts; \
                    int   _ti = I_[_j]; I_[_j] = I_[_j-1]; I_[_j-1] = _ti; \
                }                                             \
            }                                                 \
        }                                                     \
    } while (0)

// ---------------- fused bf16-WMMA GEMM (2x2 blocked) + per-split top-16 -------
// grid = (BATCH/QTILE, NSPLIT), block = 256 (8 wave32)
__global__ void __launch_bounds__(256)
gemm_topk_kernel(const unsigned short* __restrict__ Qb,
                 const unsigned short* __restrict__ Ib,
                 float* __restrict__ partS,
                 int*   __restrict__ partI) {
    __shared__ __align__(16) unsigned short aLds[QTILE * EMBED];   // 32 KB query tile
    __shared__ __align__(16) unsigned char  scratch[32768];        // 32 KB overlay
    float* scoreBuf = (float*)scratch;                  // [32][SB_STRIDE], per half-block
    float* candS    = (float*)scratch;                  // [32][128] (after nb loop)
    int*   candI    = (int*)(scratch + QTILE * TPQ * TOPK * 4);

    const int qbase     = blockIdx.x * QTILE;
    const int splitBase = blockIdx.y * SPLIT_CHUNKS;

    // ---- stage Q tile (32 x 512 bf16 = 32 KB) into LDS ----
    {
        const uint4* srcv = (const uint4*)(Qb + (size_t)qbase * EMBED);
        uint4* dstv = (uint4*)aLds;
        #pragma unroll
        for (int i = 0; i < (QTILE * EMBED * 2 / 16) / 256; ++i)
            dstv[threadIdx.x + i * 256] = srcv[threadIdx.x + i * 256];
    }
    __syncthreads();

    const int lane = threadIdx.x & 31;
    const int wave = threadIdx.x >> 5;
    const int hi   = lane >> 4;       // lane half selects K sub-block
    const int mrow = lane & 15;       // A: M row / B,C: N column

    // A fragment bases in LDS (two M-tiles), contiguous K order (k = 16*hi + s), matching B
    const bf16_t* a0Base = ((const bf16_t*)aLds) + (mrow)      * EMBED + hi * 16;
    const bf16_t* a1Base = ((const bf16_t*)aLds) + (16 + mrow) * EMBED + hi * 16;

    // private running top-16 (sorted descending)
    float tS[TOPK]; int tI[TOPK];
    TOPK_INIT(tS, tI);
    const int q   = threadIdx.x & 31; // query row this thread scans
    const int sub = threadIdx.x >> 5; // sub-scanner id (== wave)

    for (int nb = 0; nb < NB_ITERS; ++nb) {
        const int blockBase = splitBase + nb * NBLOCK;
        const int chunk0 = blockBase + wave * NPW;      // this wave's 32 chunks
        // B fragment bases (two N-tiles): column = mrow, K half by hi*16
        const bf16_t* b0Base = (const bf16_t*)Ib + (size_t)(chunk0 + mrow)      * EMBED + hi * 16;
        const bf16_t* b1Base = (const bf16_t*)Ib + (size_t)(chunk0 + 16 + mrow) * EMBED + hi * 16;

        if (nb + 1 < NB_ITERS) {
            const bf16_t* pre0 = (const bf16_t*)Ib +
                (size_t)(blockBase + NBLOCK + wave * NPW + mrow) * EMBED;
            __builtin_prefetch((const void*)pre0, 0, 1);
            __builtin_prefetch((const void*)(pre0 + (size_t)16 * EMBED), 0, 1);
        }

        v8f c00 = {}, c01 = {}, c10 = {}, c11 = {};
        #pragma unroll
        for (int kk = 0; kk < EMBED; kk += 32) {
            v8bf a0lo = *(const v8bf*)(a0Base + kk);
            v8bf a0hi = *(const v8bf*)(a0Base + kk + 8);
            v16bf A0 = __builtin_shufflevector(a0lo, a0hi,
                         0,1,2,3,4,5,6,7,8,9,10,11,12,13,14,15);
            v8bf a1lo = *(const v8bf*)(a1Base + kk);
            v8bf a1hi = *(const v8bf*)(a1Base + kk + 8);
            v16bf A1 = __builtin_shufflevector(a1lo, a1hi,
                         0,1,2,3,4,5,6,7,8,9,10,11,12,13,14,15);
            v16bf B0 = *(const v16bf*)(b0Base + kk);
            v16bf B1 = *(const v16bf*)(b1Base + kk);
            // 4 independent accumulator chains -> pipelined matrix unit
            c00 = __builtin_amdgcn_wmma_f32_16x16x32_bf16(false, A0, false, B0, (short)0, c00, false, false);
            c01 = __builtin_amdgcn_wmma_f32_16x16x32_bf16(false, A0, false, B1, (short)0, c01, false, false);
            c10 = __builtin_amdgcn_wmma_f32_16x16x32_bf16(false, A1, false, B0, (short)0, c10, false, false);
            c11 = __builtin_amdgcn_wmma_f32_16x16x32_bf16(false, A1, false, B1, (short)0, c11, false, false);
        }

        // ---- two half-block rounds: ni=0 -> {c00,c10}, ni=1 -> {c01,c11} ----
        #pragma unroll
        for (int ni = 0; ni < 2; ++ni) {
            const v8f* f0 = ni ? &c01 : &c00;
            const v8f* f1 = ni ? &c11 : &c10;
            // C layout: VGPR r -> row r + 8*hi, col = mrow; LDS col = wave*16+mrow
            #pragma unroll
            for (int r = 0; r < 8; ++r) {
                scoreBuf[(r + hi * 8)      * SB_STRIDE + wave * 16 + mrow] = (*f0)[r];
                scoreBuf[(16 + r + hi * 8) * SB_STRIDE + wave * 16 + mrow] = (*f1)[r];
            }
            __syncthreads();
            // 8 threads per query scan 128 fresh scores (16 each)
            #pragma unroll
            for (int c8 = 0; c8 < 128 / TPQ; ++c8) {
                int col = c8 * TPQ + sub;
                float v = scoreBuf[q * SB_STRIDE + col];
                int chunk = blockBase + (col >> 4) * NPW + ni * 16 + (col & 15);
                TOPK_INSERT(tS, tI, v, chunk);
            }
            __syncthreads();
        }
    }

    // dump 16 candidates per thread -> 128 per query, reduce to 16
    #pragma unroll
    for (int j = 0; j < TOPK; ++j) {
        candS[q * (TPQ * TOPK) + sub * TOPK + j] = tS[j];
        candI[q * (TPQ * TOPK) + sub * TOPK + j] = tI[j];
    }
    __syncthreads();

    if (threadIdx.x < QTILE) {
        const int q2 = threadIdx.x;
        float bS[TOPK]; int bI[TOPK];
        TOPK_INIT(bS, bI);
        for (int i = 0; i < TPQ * TOPK; ++i)
            TOPK_INSERT(bS, bI, candS[q2 * (TPQ * TOPK) + i], candI[q2 * (TPQ * TOPK) + i]);
        size_t base = ((size_t)(qbase + q2) * NSPLIT + blockIdx.y) * TOPK;
        #pragma unroll
        for (int j = 0; j < TOPK; ++j) {
            partS[base + j] = bS[j];
            partI[base + j] = bI[j];
        }
    }
}

// ---------------- merge per-split candidates, gather positions ----------------
__global__ void __launch_bounds__(256)
merge_topk_kernel(const float* __restrict__ partS,
                  const int*   __restrict__ partI,
                  const int*   __restrict__ positions,
                  float* __restrict__ outS,
                  int*   __restrict__ outP) {
    int qq = blockIdx.x * blockDim.x + threadIdx.x;
    if (qq >= BATCH) return;
    float bS[TOPK]; int bI[TOPK];
    TOPK_INIT(bS, bI);
    size_t base = (size_t)qq * NSPLIT * TOPK;
    for (int i = 0; i < NSPLIT * TOPK; ++i)
        TOPK_INSERT(bS, bI, partS[base + i], partI[base + i]);
    #pragma unroll
    for (int j = 0; j < TOPK; ++j) {
        outS[(size_t)qq * TOPK + j] = bS[j];
        outP[(size_t)qq * TOPK + j] = positions[bI[j]];
    }
}

// ---------------- launcher ----------------
extern "C" void kernel_launch(void* const* d_in, const int* in_sizes, int n_in,
                              void* d_out, int out_size, void* d_ws, size_t ws_size,
                              hipStream_t stream) {
    (void)in_sizes; (void)n_in; (void)out_size; (void)ws_size;
    const float* Q   = (const float*)d_in[0];
    const float* I   = (const float*)d_in[1];
    const int*   pos = (const int*)d_in[2];

    char* ws = (char*)d_ws;
    const size_t QB_BYTES = (size_t)BATCH * EMBED * 2;        // 2 MB
    const size_t IB_BYTES = (size_t)NUM_CHUNKS * EMBED * 2;   // 64 MB
    const size_t PS_BYTES = (size_t)BATCH * NSPLIT * TOPK * 4;// 8 MB
    unsigned short* Qb = (unsigned short*)(ws);
    unsigned short* Ib = (unsigned short*)(ws + ((QB_BYTES + 255) & ~(size_t)255));
    float* partS = (float*)(ws + ((QB_BYTES + 255) & ~(size_t)255)
                               + ((IB_BYTES + 255) & ~(size_t)255));
    int*   partI = (int*)((char*)partS + PS_BYTES);

    int nq4 = BATCH * EMBED / 4;
    int ni4 = NUM_CHUNKS * EMBED / 4;
    cvt_bf16_kernel<<<nq4 / 256, 256, 0, stream>>>(Q, Qb, nq4);
    cvt_bf16_kernel<<<ni4 / 256, 256, 0, stream>>>(I, Ib, ni4);

    dim3 grid(BATCH / QTILE, NSPLIT);
    gemm_topk_kernel<<<grid, 256, 0, stream>>>(Qb, Ib, partS, partI);

    float* outS = (float*)d_out;
    int*   outP = (int*)d_out + (size_t)BATCH * TOPK;
    merge_topk_kernel<<<BATCH / 256, 256, 0, stream>>>(partS, partI, pos, outS, outP);
}
